// CausalSelfAttention_88716844466366
// MI455X (gfx1250) — matrix operand order repaired and checked by reading, not compiled, not run on hardware
//
#include <hip/hip_runtime.h>

// Problem constants (from reference)
#define N_EMBD   1024
#define N_HEAD   16
#define HEAD_DIM 64
#define BATCH    4
#define SEQ      2048
#define MTOT     (BATCH * SEQ)   // 8192 rows

typedef __attribute__((ext_vector_type(16))) _Float16 v16h;
typedef __attribute__((ext_vector_type(8)))  float    v8f;

struct U32B { uint4 a, b; };   // 32 bytes -> bit_cast to v16h

// A-operand (16x32 f16) per ISA table: lane r=lane&15 is row M; half=lane>>4.
// VGPR0..3 = K[k0 + half*8 .. +7], VGPR4..7 = K[k0+16+half*8 .. +7]
__device__ __forceinline__ v16h ldA(const _Float16* p /* row base + k0 */, int half) {
    U32B u;
    u.a = *reinterpret_cast<const uint4*>(p + half * 8);
    u.b = *reinterpret_cast<const uint4*>(p + 16 + half * 8);
    return __builtin_bit_cast(v16h, u);
}

// B-operand (32x16 f16): lane = column N; same K order as A:
// VGPR0..3 = K[k0 + half*8 .. +7], VGPR4..7 = K[k0+16+half*8 .. +7]. Storage is
// N-major, so these are two 16-byte reads.
__device__ __forceinline__ v16h ldB(const _Float16* p /* col-row base + k0 */, int half) {
    U32B u;
    u.a = *reinterpret_cast<const uint4*>(p + half * 8);
    u.b = *reinterpret_cast<const uint4*>(p + 16 + half * 8);
    return __builtin_bit_cast(v16h, u);
}

__device__ __forceinline__ v8f wmma16(v16h a, v16h b, v8f c) {
    return __builtin_amdgcn_wmma_f32_16x16x32_f16(false, a, false, b, (short)0, c, false, false);
}

__device__ __forceinline__ float rmax16(float v) {
#pragma unroll
    for (int i = 1; i < 16; i <<= 1) v = fmaxf(v, __shfl_xor(v, i, 32));
    return v;
}
__device__ __forceinline__ float rsum16(float v) {
#pragma unroll
    for (int i = 1; i < 16; i <<= 1) v += __shfl_xor(v, i, 32);
    return v;
}

// ---------------------------------------------------------------------------
// Tensor Data Mover (gfx1250). D# per ISA 08_async_tensor §8.3/8.4.
// Loads a tile_w x tile_h (elements x rows) f16 tile; pad_enable inserts
// pad_amount dwords after every pad_interval dwords so LDS rows land at a
// bank-conflict-free, 16B-aligned pitch.
// ---------------------------------------------------------------------------
#if __has_builtin(__builtin_amdgcn_tensor_load_to_lds)
#define USE_TDM 1
typedef __attribute__((ext_vector_type(4))) unsigned int u32x4;
typedef __attribute__((ext_vector_type(8))) int i32x8;
typedef __attribute__((ext_vector_type(4))) int i32x4;

__device__ __forceinline__ unsigned lds_off(const void* p) {
    return (unsigned)(unsigned long)(__attribute__((address_space(3))) const void*)p;
}

__device__ __forceinline__ void tdm_load_tile(unsigned lds_addr, const _Float16* g,
                                              int strideElems, int tile_w, int tile_h,
                                              unsigned pad_interval_code,
                                              unsigned pad_amount_code) {
    unsigned long ga = (unsigned long)g;
    u32x4 g0;
    g0.x = 1u;                                   // count=1, user descriptor
    g0.y = lds_addr;                             // lds_addr (bytes)
    g0.z = (unsigned)ga;                         // global_addr[31:0]
    g0.w = (unsigned)((ga >> 32) & 0x01ffffffu) | (2u << 30);  // addr[56:32] | type=2
    i32x8 g1;
    // data_size=1 (2B) | pad_enable | pad_interval | pad_amount
    g1[0] = (int)((1u << 16) | (1u << 20) | (pad_interval_code << 22) |
                  (pad_amount_code << 25));
    g1[1] = (int)((unsigned)tile_w << 16);       // tensor_dim0[15:0] in [31:16]
    g1[2] = (int)((unsigned)tile_h << 16);       // tensor_dim0 hi=0 | tensor_dim1 lo
    g1[3] = (int)((unsigned)tile_w << 16);       // tensor_dim1 hi=0 | tile_dim0
    g1[4] = tile_h;                              // tile_dim1, tile_dim2=0
    g1[5] = strideElems;                         // tensor_dim0_stride[31:0]
    g1[6] = 0;
    g1[7] = 0;
    i32x4 z4 = {0, 0, 0, 0};
    i32x8 z8 = {0, 0, 0, 0, 0, 0, 0, 0};
    // 6-arg form (clang-23 / therock-10.0): (g0, g1, g2, g3, g4, cpol)
    __builtin_amdgcn_tensor_load_to_lds(g0, g1, z4, z4, z8, 0);
}
#else
#define USE_TDM 0
#endif

#if __has_builtin(__builtin_amdgcn_s_wait_tensorcnt)
#define WAIT_TENSOR(n) __builtin_amdgcn_s_wait_tensorcnt(n)
#else
#define WAIT_TENSOR(n) asm volatile("s_wait_tensorcnt %0" ::"i"(n))
#endif

// ---------------------------------------------------------------------------
// Shared block-GEMM core: 128(M) x 128(N) tile, K-step 32, 8 waves of 64x32.
// A row-major [M][K]; B N-major [N][K]. LDS rows at 80B pitch (40 halves).
// ---------------------------------------------------------------------------
#define LDS_PITCH 40            // halves per LDS row (80 bytes)
#define BUF_HALVES (128 * LDS_PITCH)

__device__ __forceinline__ void gemm_block_128x128(const _Float16* __restrict__ Aglob,
                                                   const _Float16* __restrict__ BglobT,
                                                   int K, int M0, int N0,
                                                   _Float16* Abuf, _Float16* Bbuf,
                                                   v8f (&acc)[4][2]) {
    const int tid = threadIdx.x;
    const int wv = tid >> 5, lane = tid & 31, half = lane >> 4, lr = lane & 15;
    const int wm = wv >> 2, wn = wv & 3;
    const int NK = K / 32;
#pragma unroll
    for (int mt = 0; mt < 4; mt++)
#pragma unroll
        for (int nt = 0; nt < 2; nt++) acc[mt][nt] = v8f{};

#if USE_TDM
    unsigned ldsA[2] = {lds_off(Abuf), lds_off(Abuf + BUF_HALVES)};
    unsigned ldsB[2] = {lds_off(Bbuf), lds_off(Bbuf + BUF_HALVES)};
    if (wv == 0) {
        tdm_load_tile(ldsA[0], Aglob + (long)M0 * K, K, 32, 128, 3, 3);
        tdm_load_tile(ldsB[0], BglobT + (long)N0 * K, K, 32, 128, 3, 3);
    }
    for (int i = 0; i < NK; i++) {
        int cur = i & 1;
        if (wv == 0) {
            if (i + 1 < NK) {
                tdm_load_tile(ldsA[cur ^ 1], Aglob + (long)M0 * K + (i + 1) * 32, K, 32, 128, 3, 3);
                tdm_load_tile(ldsB[cur ^ 1], BglobT + (long)N0 * K + (i + 1) * 32, K, 32, 128, 3, 3);
                WAIT_TENSOR(2);   // the 2 older TDM ops (buffer `cur`) are done
            } else {
                WAIT_TENSOR(0);
            }
        }
        __syncthreads();          // publish buffer `cur`
        const _Float16* Ab = Abuf + cur * BUF_HALVES;
        const _Float16* Bb = Bbuf + cur * BUF_HALVES;
        v16h a[4], bb[2];
#pragma unroll
        for (int mt = 0; mt < 4; mt++)
            a[mt] = ldA(Ab + (wm * 64 + mt * 16 + lr) * LDS_PITCH, half);
#pragma unroll
        for (int nt = 0; nt < 2; nt++)
            bb[nt] = ldB(Bb + (wn * 32 + nt * 16 + lr) * LDS_PITCH, half);
#pragma unroll
        for (int mt = 0; mt < 4; mt++)
#pragma unroll
            for (int nt = 0; nt < 2; nt++) acc[mt][nt] = wmma16(a[mt], bb[nt], acc[mt][nt]);
        __syncthreads();          // done reading before overwrite
    }
#else
    for (int i = 0; i < NK; i++) {
        int k0 = i * 32;
        for (int c = tid; c < 1024; c += blockDim.x) {
            int mat = c >> 9, idx = c & 511;
            int row = idx >> 2, sub = idx & 3;
            const _Float16* g = (mat ? BglobT + (long)(N0 + row) * K
                                     : Aglob + (long)(M0 + row) * K) + k0 + sub * 8;
            uint4 v = *reinterpret_cast<const uint4*>(g);
            _Float16* d = (mat ? Bbuf : Abuf) + row * LDS_PITCH + sub * 8;
            *reinterpret_cast<uint4*>(d) = v;
        }
        __syncthreads();
        v16h a[4], bb[2];
#pragma unroll
        for (int mt = 0; mt < 4; mt++)
            a[mt] = ldA(Abuf + (wm * 64 + mt * 16 + lr) * LDS_PITCH, half);
#pragma unroll
        for (int nt = 0; nt < 2; nt++)
            bb[nt] = ldB(Bbuf + (wn * 32 + nt * 16 + lr) * LDS_PITCH, half);
#pragma unroll
        for (int mt = 0; mt < 4; mt++)
#pragma unroll
            for (int nt = 0; nt < 2; nt++) acc[mt][nt] = wmma16(a[mt], bb[nt], acc[mt][nt]);
        __syncthreads();
    }
#endif
}

// ---------------------------------------------------------------------------
// K0: fp32 -> f16 conversion; transpose weights to N-major for B-operand
// ---------------------------------------------------------------------------
__global__ void k_convert(const float* __restrict__ x, const float* __restrict__ wqkv,
                          const float* __restrict__ wout,
                          _Float16* __restrict__ xb, _Float16* __restrict__ wqkvT,
                          _Float16* __restrict__ woutT) {
    const long NX  = (long)MTOT * N_EMBD;
    const long NW1 = (long)N_EMBD * 3 * N_EMBD;
    const long NW2 = (long)N_EMBD * N_EMBD;
    const long total = NX + NW1 + NW2;
    for (long i = (long)blockIdx.x * blockDim.x + threadIdx.x; i < total;
         i += (long)gridDim.x * blockDim.x) {
        if (i < NX) {
            xb[i] = (_Float16)x[i];
        } else if (i < NX + NW1) {
            long j = i - NX;
            int n = (int)(j / N_EMBD), k = (int)(j % N_EMBD);
            wqkvT[j] = (_Float16)wqkv[(long)k * (3 * N_EMBD) + n];
        } else {
            long j = i - NX - NW1;
            int n = (int)(j / N_EMBD), k = (int)(j % N_EMBD);
            woutT[j] = (_Float16)wout[(long)k * N_EMBD + n];
        }
    }
}

// ---------------------------------------------------------------------------
// K1: QKV projection GEMM (M=8192, N=3072, K=1024). Block tile 128x128.
// Epilogue scatters Q,K row-major [B,H,T,D] and V transposed [B,H,D,T].
// ---------------------------------------------------------------------------
__global__ void __launch_bounds__(256) k_qkv(const _Float16* __restrict__ xb,
                                             const _Float16* __restrict__ wT,
                                             const float* __restrict__ bias,
                                             _Float16* __restrict__ Qb,
                                             _Float16* __restrict__ Kb,
                                             _Float16* __restrict__ VTb) {
    __shared__ __align__(16) _Float16 Abuf[2 * BUF_HALVES];
    __shared__ __align__(16) _Float16 Bbuf[2 * BUF_HALVES];
    const int wv = threadIdx.x >> 5, lane = threadIdx.x & 31;
    const int half = lane >> 4, lr = lane & 15, wm = wv >> 2, wn = wv & 3;
    const int MT = MTOT / 128, NT = (3 * N_EMBD) / 128;   // 64 x 24 block tiles
    for (int job = blockIdx.x; job < MT * NT; job += gridDim.x) {
        int M0 = (job % MT) * 128, N0 = (job / MT) * 128;
        v8f acc[4][2];
        gemm_block_128x128(xb, wT, N_EMBD, M0, N0, Abuf, Bbuf, acc);
#pragma unroll
        for (int mt = 0; mt < 4; mt++) {
#pragma unroll
            for (int nt = 0; nt < 2; nt++) {
                int n = N0 + wn * 32 + nt * 16 + lr;
                float bv = bias[n];
                int region = n >> 10, nn = n & 1023, h = nn >> 6, d = nn & 63;
#pragma unroll
                for (int j = 0; j < 8; j++) {
                    int m = M0 + wm * 64 + mt * 16 + j + half * 8;
                    int b = m >> 11, t = m & (SEQ - 1);
                    _Float16 hv = (_Float16)(acc[mt][nt][j] + bv);
                    long hb = (long)(b * N_HEAD + h);
                    if (region == 0)      Qb[(hb * SEQ + t) * HEAD_DIM + d] = hv;
                    else if (region == 1) Kb[(hb * SEQ + t) * HEAD_DIM + d] = hv;
                    else                  VTb[(hb * HEAD_DIM + d) * SEQ + t] = hv;
                }
            }
        }
    }
}

// ---------------------------------------------------------------------------
// K2: Workgroup-cooperative flash attention. One block = 128 query rows of
// one (b,h); 8 waves of 16 rows. K/V 32-key tiles TDM-staged into LDS and
// shared by all waves (8x traffic reduction), double-buffered.
//   K tile: 32 rows x 64 d, pitch 144B (pad 4dw/32dw)   -> S = Q K^T B-operand
//   V tile: 64 d-rows x 32 s, pitch 80B (pad 4dw/16dw)  -> Y = P V  B-operand
// ---------------------------------------------------------------------------
#define KT_PITCH 72             // halves (144 B)
#define KT_HALVES (32 * KT_PITCH)
#define VT_PITCH 40             // halves (80 B)
#define VT_HALVES (64 * VT_PITCH)

__global__ void __launch_bounds__(256) k_attn(const _Float16* __restrict__ Qb,
                                              const _Float16* __restrict__ Kb,
                                              const _Float16* __restrict__ VTb,
                                              _Float16* __restrict__ Yb) {
    __shared__ __align__(16) _Float16 Kt[2][KT_HALVES];
    __shared__ __align__(16) _Float16 Vs[2][VT_HALVES];
    __shared__ __align__(16) _Float16 pbuf[8][16][40];
    const int wv = threadIdx.x >> 5, lane = threadIdx.x & 31;
    const int half = lane >> 4, lr = lane & 15;
    const int QB = SEQ / 128;                       // 16 q-blocks per head
    const int njobs = BATCH * N_HEAD * QB;          // 1024
    const float NEGINF = -__builtin_inff();
    for (int job = blockIdx.x; job < njobs; job += gridDim.x) {
        int qb = job % QB, h = (job / QB) % N_HEAD, b = job / (QB * N_HEAD);
        int Qb0 = qb * 128;
        int Q0  = Qb0 + wv * 16;                    // this wave's query rows
        long hbase = (long)(b * N_HEAD + h) * SEQ * HEAD_DIM;
        const _Float16* Qh = Qb + hbase;
        const _Float16* Kh = Kb + hbase;
        const _Float16* Vh = VTb + hbase;           // [D][T] per head

        v16h aq0 = ldA(Qh + (long)(Q0 + lr) * HEAD_DIM + 0, half);
        v16h aq1 = ldA(Qh + (long)(Q0 + lr) * HEAD_DIM + 32, half);
        v8f y0 = {}, y1 = {}, y2 = {}, y3 = {};
        float mrow[8], lrow[8];
#pragma unroll
        for (int j = 0; j < 8; j++) { mrow[j] = NEGINF; lrow[j] = 0.f; }

        const int nkb = (Qb0 + 128) / 32;           // key blocks for whole q-block
#if USE_TDM
        if (wv == 0) {
            tdm_load_tile(lds_off(&Kt[0][0]), Kh, HEAD_DIM, 64, 32, 4, 3);
            tdm_load_tile(lds_off(&Vs[0][0]), Vh, SEQ, 32, 64, 3, 3);
        }
#endif
        for (int i = 0; i < nkb; i++) {
            int S0 = i * 32;
#if USE_TDM
            int cur = i & 1;
            if (wv == 0) {
                if (i + 1 < nkb) {
                    tdm_load_tile(lds_off(&Kt[cur ^ 1][0]), Kh + (long)(S0 + 32) * HEAD_DIM,
                                  HEAD_DIM, 64, 32, 4, 3);
                    tdm_load_tile(lds_off(&Vs[cur ^ 1][0]), Vh + S0 + 32, SEQ, 32, 64, 3, 3);
                    WAIT_TENSOR(2);
                } else {
                    WAIT_TENSOR(0);
                }
            }
            __syncthreads();
#else
            int cur = 0;
            {   // synchronous cooperative staging
                int row = threadIdx.x >> 3, sub = threadIdx.x & 7;   // K: 32 rows x 8 chunks
                *reinterpret_cast<uint4*>(&Kt[0][row * KT_PITCH + sub * 8]) =
                    *reinterpret_cast<const uint4*>(Kh + (long)(S0 + row) * HEAD_DIM + sub * 8);
                int vrow = threadIdx.x >> 2, vsub = threadIdx.x & 3; // V: 64 rows x 4 chunks
                *reinterpret_cast<uint4*>(&Vs[0][vrow * VT_PITCH + vsub * 8]) =
                    *reinterpret_cast<const uint4*>(Vh + (long)vrow * SEQ + S0 + vsub * 8);
            }
            __syncthreads();
#endif
            if (S0 < Q0 + 16) {   // causal range of this wave (uniform per wave)
                v8f s0v = {}, s1v = {};
                const _Float16* Kr = &Kt[cur][0];
                s0v = wmma16(aq0, ldB(Kr + lr * KT_PITCH + 0, half), s0v);
                s0v = wmma16(aq1, ldB(Kr + lr * KT_PITCH + 32, half), s0v);
                s1v = wmma16(aq0, ldB(Kr + (16 + lr) * KT_PITCH + 0, half), s1v);
                s1v = wmma16(aq1, ldB(Kr + (16 + lr) * KT_PITCH + 32, half), s1v);

                const float sc = 0.125f;            // 1/sqrt(64)
                bool need_mask = (S0 + 31 > Q0);
                int rbase = Q0 + half * 8;
                float scl[8];
#pragma unroll
                for (int j = 0; j < 8; j++) {
                    float v0 = s0v[j] * sc, v1 = s1v[j] * sc;
                    if (need_mask) {
                        int t = rbase + j;
                        if (S0 + lr > t)      v0 = NEGINF;
                        if (S0 + 16 + lr > t) v1 = NEGINF;
                    }
                    float mx = rmax16(fmaxf(v0, v1));
                    float mn = fmaxf(mrow[j], mx);
                    float p0 = __expf(v0 - mn), p1 = __expf(v1 - mn);
                    float rs = rsum16(p0 + p1);
                    scl[j]  = __expf(mrow[j] - mn);
                    lrow[j] = lrow[j] * scl[j] + rs;
                    mrow[j] = mn;
                    int r = j + half * 8;           // C/D-layout row for VGPR j
                    pbuf[wv][r][lr]      = (_Float16)p0;
                    pbuf[wv][r][16 + lr] = (_Float16)p1;
                }
#pragma unroll
                for (int j = 0; j < 8; j++) {
                    y0[j] *= scl[j]; y1[j] *= scl[j]; y2[j] *= scl[j]; y3[j] *= scl[j];
                }
                v16h ap = ldA(&pbuf[wv][lr][0], half);
                const _Float16* Vr = &Vs[cur][0];
                y0 = wmma16(ap, ldB(Vr + (0 * 16 + lr) * VT_PITCH, half), y0);
                y1 = wmma16(ap, ldB(Vr + (1 * 16 + lr) * VT_PITCH, half), y1);
                y2 = wmma16(ap, ldB(Vr + (2 * 16 + lr) * VT_PITCH, half), y2);
                y3 = wmma16(ap, ldB(Vr + (3 * 16 + lr) * VT_PITCH, half), y3);
            }
            __syncthreads();      // all waves done reading before next TDM overwrite
        }
        // Normalize and store Y row-major [B*T][N_EMBD]
#pragma unroll
        for (int j = 0; j < 8; j++) {
            int t = Q0 + j + half * 8;
            float inv = 1.0f / lrow[j];
            long row = ((long)b * SEQ + t) * N_EMBD + h * HEAD_DIM;
            Yb[row + 0 * 16 + lr] = (_Float16)(y0[j] * inv);
            Yb[row + 1 * 16 + lr] = (_Float16)(y1[j] * inv);
            Yb[row + 2 * 16 + lr] = (_Float16)(y2[j] * inv);
            Yb[row + 3 * 16 + lr] = (_Float16)(y3[j] * inv);
        }
    }
}

// ---------------------------------------------------------------------------
// K3: Output projection (M=8192, N=1024, K=1024). Block tile 128x128, f32 out.
// ---------------------------------------------------------------------------
__global__ void __launch_bounds__(256) k_out(const _Float16* __restrict__ Yb,
                                             const _Float16* __restrict__ wT,
                                             const float* __restrict__ bias,
                                             float* __restrict__ out) {
    __shared__ __align__(16) _Float16 Abuf[2 * BUF_HALVES];
    __shared__ __align__(16) _Float16 Bbuf[2 * BUF_HALVES];
    const int wv = threadIdx.x >> 5, lane = threadIdx.x & 31;
    const int half = lane >> 4, lr = lane & 15, wm = wv >> 2, wn = wv & 3;
    const int MT = MTOT / 128, NT = N_EMBD / 128;   // 64 x 8 block tiles
    for (int job = blockIdx.x; job < MT * NT; job += gridDim.x) {
        int M0 = (job % MT) * 128, N0 = (job / MT) * 128;
        v8f acc[4][2];
        gemm_block_128x128(Yb, wT, N_EMBD, M0, N0, Abuf, Bbuf, acc);
#pragma unroll
        for (int mt = 0; mt < 4; mt++) {
#pragma unroll
            for (int nt = 0; nt < 2; nt++) {
                int n = N0 + wn * 32 + nt * 16 + lr;
                float bv = bias[n];
#pragma unroll
                for (int j = 0; j < 8; j++) {
                    int m = M0 + wm * 64 + mt * 16 + j + half * 8;
                    out[(long)m * N_EMBD + n] = acc[mt][nt][j] + bv;
                }
            }
        }
    }
}

// ---------------------------------------------------------------------------
// Workspace layout (bytes), total 92,274,688 required:
//  xb@0 (16MB)  wqkvT@16777216 (6MB)  woutT@23068672 (2MB)
//  Qb@25165824  Kb@41943040  VTb@58720256  Yb@75497472   (16MB each)
// ---------------------------------------------------------------------------
extern "C" void kernel_launch(void* const* d_in, const int* in_sizes, int n_in,
                              void* d_out, int out_size, void* d_ws, size_t ws_size,
                              hipStream_t stream) {
    const float* x    = (const float*)d_in[0];
    const float* wqkv = (const float*)d_in[1];
    const float* bqkv = (const float*)d_in[2];
    const float* wout = (const float*)d_in[3];
    const float* bout = (const float*)d_in[4];
    float* out = (float*)d_out;

    char* ws = (char*)d_ws;
    _Float16* xb    = (_Float16*)(ws + 0);
    _Float16* wqkvT = (_Float16*)(ws + 16777216L);
    _Float16* woutT = (_Float16*)(ws + 23068672L);
    _Float16* Qb    = (_Float16*)(ws + 25165824L);
    _Float16* Kb    = (_Float16*)(ws + 41943040L);
    _Float16* VTb   = (_Float16*)(ws + 58720256L);
    _Float16* Yb    = (_Float16*)(ws + 75497472L);

    k_convert<<<4096, 256, 0, stream>>>(x, wqkv, wout, xb, wqkvT, woutT);
    k_qkv<<<1536, 256, 0, stream>>>(xb, wqkvT, bqkv, Qb, Kb, VTb);   // 64x24 block tiles
    k_attn<<<1024, 256, 0, stream>>>(Qb, Kb, VTb, Yb);               // 1024 q-block jobs
    k_out<<<512, 256, 0, stream>>>(Yb, woutT, bout, out);            // 64x8 block tiles
}
